// TreeBasedConvolutionLayer_88227218194540
// MI455X (gfx1250) — matrix operand order repaired and checked by reading, not compiled, not run on hardware
//
#include <hip/hip_runtime.h>
#include <math.h>

// ---------------------------------------------------------------------------
// Tree-based convolution, restructured as:
//   S0[s] = sum_{n in window(s)} X[n]          (window = self + children)
//   S2[s] = sum_{n in window(s)} r_n * X[n],   r_n = (pos-1)/max(1,sib-1)
//   out   = tanh( twc*(S0 Wt^T) + w*(S0 Wl^T) + w*(S2 Wr^T) - w^2*(S2 Wl^T)
//                 + cnt*bias ),  twc = level/2, w = 1-twc
// GEMMs on v_wmma_f32_16x16x32_bf16; A-slabs staged fp32 -> LDS via
// double-buffered async global->LDS copies (ASYNCcnt) when available.
// ---------------------------------------------------------------------------

#define FDIM 512
#define CDIM 512
#define BM   32      // output rows per block
#define KB   32      // K step per WMMA
#define LDAF 36      // padded LDS row stride in floats (conflict-free b128)

typedef __bf16 bf16;
typedef __bf16 bf16x16 __attribute__((ext_vector_type(16)));
typedef float  v8f     __attribute__((ext_vector_type(8)));
typedef int    v4i     __attribute__((vector_size(16)));   // 16B chunk type

#if defined(__gfx1250__) && __has_builtin(__builtin_amdgcn_global_load_async_to_lds_b128)
#define HAVE_ASYNC 1
#endif

#ifdef HAVE_ASYNC
__device__ __forceinline__ void async_cp16(const float* g, float* l) {
    __builtin_amdgcn_global_load_async_to_lds_b128(
        (v4i __attribute__((address_space(1)))*)(void*)g,
        (v4i __attribute__((address_space(3)))*)(void*)l, 0, 0);
}
__device__ __forceinline__ void async_wait0() {
#if __has_builtin(__builtin_amdgcn_s_wait_asynccnt)
    __builtin_amdgcn_s_wait_asynccnt(0);
#else
    asm volatile("s_wait_asynccnt 0" ::: "memory");
#endif
}
#else
__device__ __forceinline__ void async_cp16(const float* g, float* l) {
    *(float4*)l = *(const float4*)g;   // sync fallback: load + ds_store
}
__device__ __forceinline__ void async_wait0() {}
#endif

__device__ __forceinline__ float node_r(const int* pos, const int* sib, int n) {
    int sd = sib[n] - 1;
    return (float)(pos[n] - 1) / (float)(sd < 1 ? 1 : sd);
}

// build a 16x32-layout bf16 A fragment from 16 fp32 LDS values (2 chunks of 8)
__device__ __forceinline__ bf16x16 cvt_frag(const float* p0, const float* p1) {
    const float4 f0 = *(const float4*)(p0);
    const float4 f1 = *(const float4*)(p0 + 4);
    const float4 f2 = *(const float4*)(p1);
    const float4 f3 = *(const float4*)(p1 + 4);
    bf16x16 r = {(bf16)f0.x, (bf16)f0.y, (bf16)f0.z, (bf16)f0.w,
                 (bf16)f1.x, (bf16)f1.y, (bf16)f1.z, (bf16)f1.w,
                 (bf16)f2.x, (bf16)f2.y, (bf16)f2.z, (bf16)f2.w,
                 (bf16)f3.x, (bf16)f3.y, (bf16)f3.z, (bf16)f3.w};
    return r;
}

// --- phase 1: self-edge init (plain stores; no zero pass required) ----------
__global__ void k_init(const float* __restrict__ X, const int* __restrict__ pos,
                       const int* __restrict__ sib, float* __restrict__ S0,
                       float* __restrict__ S2, float* __restrict__ cntF, int N) {
    const size_t total = (size_t)N * (FDIM / 4);
    for (size_t i = (size_t)blockIdx.x * blockDim.x + threadIdx.x; i < total;
         i += (size_t)gridDim.x * blockDim.x) {
        const int n  = (int)(i / (FDIM / 4));
        const int f4 = (int)(i % (FDIM / 4));
        const float r = node_r(pos, sib, n);
        const float4 x = ((const float4*)X)[i];
        ((float4*)S0)[i] = x;
        float4 s2; s2.x = r * x.x; s2.y = r * x.y; s2.z = r * x.z; s2.w = r * x.w;
        ((float4*)S2)[i] = s2;
        if (f4 == 0) cntF[n] = 1.0f;
    }
}

// --- phase 2: child-edge scatter-add into parent rows (fp32 HW atomics) -----
__global__ void k_scatter(const float* __restrict__ X, const int* __restrict__ parent,
                          const int* __restrict__ pos, const int* __restrict__ sib,
                          float* __restrict__ S0, float* __restrict__ S2,
                          float* __restrict__ cntF, int N) {
    const size_t total = (size_t)(N - 1) * (FDIM / 4);
    for (size_t i = (size_t)blockIdx.x * blockDim.x + threadIdx.x; i < total;
         i += (size_t)gridDim.x * blockDim.x) {
        const int n  = 1 + (int)(i / (FDIM / 4));
        const int f4 = (int)(i % (FDIM / 4));
        const int p  = parent[n];
        const float r = node_r(pos, sib, n);
        const float4 x = ((const float4*)X)[(size_t)n * (FDIM / 4) + f4];
        float* d0 = S0 + (size_t)p * FDIM + f4 * 4;
        float* d2 = S2 + (size_t)p * FDIM + f4 * 4;
        __hip_atomic_fetch_add(d0 + 0, x.x, __ATOMIC_RELAXED, __HIP_MEMORY_SCOPE_AGENT);
        __hip_atomic_fetch_add(d0 + 1, x.y, __ATOMIC_RELAXED, __HIP_MEMORY_SCOPE_AGENT);
        __hip_atomic_fetch_add(d0 + 2, x.z, __ATOMIC_RELAXED, __HIP_MEMORY_SCOPE_AGENT);
        __hip_atomic_fetch_add(d0 + 3, x.w, __ATOMIC_RELAXED, __HIP_MEMORY_SCOPE_AGENT);
        __hip_atomic_fetch_add(d2 + 0, r * x.x, __ATOMIC_RELAXED, __HIP_MEMORY_SCOPE_AGENT);
        __hip_atomic_fetch_add(d2 + 1, r * x.y, __ATOMIC_RELAXED, __HIP_MEMORY_SCOPE_AGENT);
        __hip_atomic_fetch_add(d2 + 2, r * x.z, __ATOMIC_RELAXED, __HIP_MEMORY_SCOPE_AGENT);
        __hip_atomic_fetch_add(d2 + 3, r * x.w, __ATOMIC_RELAXED, __HIP_MEMORY_SCOPE_AGENT);
        if (f4 == 0)
            __hip_atomic_fetch_add(&cntF[p], 1.0f, __ATOMIC_RELAXED, __HIP_MEMORY_SCOPE_AGENT);
    }
}

// --- phase 3: weights -> bf16 (done once; 1.5 MB, stays L2-resident) --------
__global__ void k_convw(const float* __restrict__ Wt_f, const float* __restrict__ Wl_f,
                        const float* __restrict__ Wr_f, bf16* __restrict__ Wt,
                        bf16* __restrict__ Wl, bf16* __restrict__ Wr) {
    const int total = CDIM * FDIM;
    for (int i = blockIdx.x * blockDim.x + threadIdx.x; i < total;
         i += gridDim.x * blockDim.x) {
        Wt[i] = (bf16)Wt_f[i];
        Wl[i] = (bf16)Wl_f[i];
        Wr[i] = (bf16)Wr_f[i];
    }
}

// --- phase 4: WMMA GEMM + coefficient epilogue + tanh -----------------------
// grid = (ceil(N/BM), CDIM/256); block = 256 threads = 8 waves.
// wave wv owns channels [blockIdx.y*256 + wv*32, +32): 2x2 subtiles of 16x16.
__global__ __launch_bounds__(256)
void k_gemm(const float* __restrict__ S0, const float* __restrict__ S2,
            const bf16* __restrict__ Wt, const bf16* __restrict__ Wl,
            const bf16* __restrict__ Wr, const float* __restrict__ bias,
            const int* __restrict__ level, const float* __restrict__ cntF,
            float* __restrict__ out, int N) {
    __shared__ float lA0[2][BM][LDAF];   // double-buffered fp32 S0 slab
    __shared__ float lA2[2][BM][LDAF];   // double-buffered fp32 S2 slab
    __shared__ float s_twc[BM];
    __shared__ float s_cnt[BM];

    const int tid  = threadIdx.x;
    const int mblk = blockIdx.x * BM;

    if (tid < BM) {
        int m = mblk + tid; if (m >= N) m = N - 1;
        s_twc[tid] = 0.5f * (float)level[m];
        s_cnt[tid] = cntF[m];
    }

    const int lane = tid & 31;
    const int wv   = tid >> 5;
    const int hs   = lane >> 4;   // half-select (K chunk / row half)
    const int lr   = lane & 15;
    const int c0w  = blockIdx.y * 256 + wv * 32;

    // 4 independent accumulator sets (no WMMA->WMMA RAW chains):
    v8f accT[2][2], accL[2][2], accR[2][2], accQ[2][2];
    #pragma unroll
    for (int i = 0; i < 2; ++i)
        #pragma unroll
        for (int j = 0; j < 2; ++j) {
            v8f z = {0.f, 0.f, 0.f, 0.f, 0.f, 0.f, 0.f, 0.f};
            accT[i][j] = z; accL[i][j] = z; accR[i][j] = z; accQ[i][j] = z;
        }

    const int sr = tid >> 3;  // staging row 0..31
    const int sc = tid & 7;   // staging 16B chunk 0..7
    int srow = mblk + sr; if (srow >= N) srow = N - 1;
    const size_t gbase = (size_t)srow * FDIM + sc * 4;

    // stage one 32x32 fp32 slab pair into LDS buffer `buf` for K offset kk
    auto stage = [&](int buf, int kk) {
        async_cp16(S0 + gbase + kk, &lA0[buf][sr][sc * 4]);
        async_cp16(S2 + gbase + kk, &lA2[buf][sr][sc * 4]);
    };

    stage(0, 0);
    int cur = 0;
    for (int kk = 0; kk < FDIM; kk += KB) {
        async_wait0();
        __syncthreads();                       // slab `cur` visible to all waves
        if (kk + KB < FDIM) stage(cur ^ 1, kk + KB);   // prefetch next slab

        // A fragments in contiguous K order (same order as B below):
        // lane holds row (lane&15); elems 0-15 = K hs*16.. (contiguous)
        bf16x16 a0f[2], a2f[2];
        #pragma unroll
        for (int ms = 0; ms < 2; ++ms) {
            const int row = ms * 16 + lr;
            a0f[ms] = cvt_frag(&lA0[cur][row][hs * 16], &lA0[cur][row][hs * 16 + 8]);
            a2f[ms] = cvt_frag(&lA2[cur][row][hs * 16], &lA2[cur][row][hs * 16 + 8]);
        }

        #pragma unroll
        for (int ns = 0; ns < 2; ++ns) {
            // B fragment: lane = column (channel), 16 contiguous K halves (32B)
            const size_t boff = (size_t)(c0w + ns * 16 + lr) * FDIM + kk + hs * 16;
            const bf16x16 bt = *(const bf16x16*)(Wt + boff);
            const bf16x16 bl = *(const bf16x16*)(Wl + boff);
            const bf16x16 br = *(const bf16x16*)(Wr + boff);
            #pragma unroll
            for (int ms = 0; ms < 2; ++ms) {
                accT[ms][ns] = __builtin_amdgcn_wmma_f32_16x16x32_bf16(
                    false, a0f[ms], false, bt, (short)0, accT[ms][ns], false, false);
                accL[ms][ns] = __builtin_amdgcn_wmma_f32_16x16x32_bf16(
                    false, a0f[ms], false, bl, (short)0, accL[ms][ns], false, false);
                accR[ms][ns] = __builtin_amdgcn_wmma_f32_16x16x32_bf16(
                    false, a2f[ms], false, br, (short)0, accR[ms][ns], false, false);
                accQ[ms][ns] = __builtin_amdgcn_wmma_f32_16x16x32_bf16(
                    false, a2f[ms], false, bl, (short)0, accQ[ms][ns], false, false);
            }
        }
        __syncthreads();                       // all reads of `cur` done
        cur ^= 1;
    }

    // epilogue: D elem v -> row (ms*16 + hs*8 + v), col (c0w + ns*16 + lr)
    #pragma unroll
    for (int ns = 0; ns < 2; ++ns) {
        const int c = c0w + ns * 16 + lr;
        const float b = bias[c];
        #pragma unroll
        for (int ms = 0; ms < 2; ++ms) {
            const v8f t = accT[ms][ns];
            const v8f l = accL[ms][ns];
            const v8f r = accR[ms][ns];
            const v8f q = accQ[ms][ns];
            #pragma unroll
            for (int v = 0; v < 8; ++v) {
                const int lm = ms * 16 + hs * 8 + v;
                const int m  = mblk + lm;
                if (m < N) {
                    const float twc = s_twc[lm];
                    const float w1  = 1.0f - twc;
                    const float val = twc * t[v] + w1 * (l[v] + r[v])
                                    - w1 * w1 * q[v] + s_cnt[lm] * b;
                    out[(size_t)m * CDIM + c] = tanhf(val);
                }
            }
        }
    }
}

// ---------------------------------------------------------------------------
extern "C" void kernel_launch(void* const* d_in, const int* in_sizes, int n_in,
                              void* d_out, int out_size, void* d_ws, size_t ws_size,
                              hipStream_t stream) {
    const float* X       = (const float*)d_in[0];
    const float* W_top   = (const float*)d_in[1];
    const float* W_left  = (const float*)d_in[2];
    const float* W_right = (const float*)d_in[3];
    const float* bias    = (const float*)d_in[4];
    const int*   parent  = (const int*)d_in[5];
    const int*   level   = (const int*)d_in[6];
    const int*   pos     = (const int*)d_in[7];
    const int*   sib     = (const int*)d_in[8];
    const int N = in_sizes[5];

    // workspace carve-out: S0, S2 (fp32 [N,F]), cntF (fp32 [N]), bf16 weights
    float* S0   = (float*)d_ws;
    float* S2   = S0 + (size_t)N * FDIM;
    float* cntF = S2 + (size_t)N * FDIM;
    size_t off  = ((size_t)(N) * FDIM * 2 + N) * sizeof(float);
    off = (off + 31) & ~(size_t)31;  // 32B align bf16 weight buffers
    bf16* Wt = (bf16*)((char*)d_ws + off);
    bf16* Wl = Wt + (size_t)CDIM * FDIM;
    bf16* Wr = Wl + (size_t)CDIM * FDIM;

    k_init   <<<2048, 256, 0, stream>>>(X, pos, sib, S0, S2, cntF, N);
    k_convw  <<<256,  256, 0, stream>>>(W_top, W_left, W_right, Wt, Wl, Wr);
    k_scatter<<<2048, 256, 0, stream>>>(X, parent, pos, sib, S0, S2, cntF, N);

    dim3 grid((N + BM - 1) / BM, CDIM / 256);
    k_gemm<<<grid, 256, 0, stream>>>(S0, S2, Wt, Wl, Wr, bias, level, cntF,
                                     (float*)d_out, N);
}
